// QwenCausalSelfAttention_28003186770425
// MI455X (gfx1250) — matrix operand order repaired and checked by reading, not compiled, not run on hardware
//
#include <hip/hip_runtime.h>
#include <hip/hip_bf16.h>

// ---------------- Problem constants (Qwen attention) ----------------
#define B_   2
#define T_   2048
#define DIM_ 896
#define H_   14
#define KV_  2
#define D_   64
#define BT_  (B_ * T_)     // 4096 token rows

// ---------------- CDNA5 WMMA types ----------------
typedef float  v8f   __attribute__((ext_vector_type(8)));
typedef __bf16 v8bf  __attribute__((ext_vector_type(8)));
typedef __bf16 v16bf __attribute__((ext_vector_type(16)));
typedef int    gv4i  __attribute__((vector_size(16)));   // 16-byte unit for async LDS copy

union BF16x16 { v16bf v; v8bf h[2]; };

#if __has_builtin(__builtin_amdgcn_global_load_async_to_lds_b128)
#define HAVE_ASYNC_LDS 1
#else
#define HAVE_ASYNC_LDS 0
#endif

__device__ __forceinline__ v8f wmma_bf16(v16bf a, v16bf b, v8f c) {
  // D = A(16x32 bf16) x B(32x16 bf16) + C(16x16 f32)
  return __builtin_amdgcn_wmma_f32_16x16x32_bf16(
      /*neg_a=*/false, a, /*neg_b=*/false, b,
      /*c_mod=*/(short)0, c, /*reuse_a=*/false, /*reuse_b=*/false);
}

// 16-byte async copy global -> LDS (ASYNCcnt path); sync fallback if builtin absent.
__device__ __forceinline__ void stage16(const __bf16* g, __bf16* l) {
#if HAVE_ASYNC_LDS
  __builtin_amdgcn_global_load_async_to_lds_b128(
      (__attribute__((address_space(1))) gv4i*)(const void*)g,
      (__attribute__((address_space(3))) gv4i*)(void*)l, 0, 0);
#else
  *(v8bf*)l = *(const v8bf*)g;
#endif
}

__device__ __forceinline__ void wait_async_lds() {
#if HAVE_ASYNC_LDS
#if __has_builtin(__builtin_amdgcn_s_wait_asynccnt)
  __builtin_amdgcn_s_wait_asynccnt(0);
#else
  asm volatile("s_wait_asynccnt 0" ::: "memory");
#endif
#endif
}

// A-fragment (16x32, 16-bit): lane L holds row (L&15); elements 0..15 -> K = kh..kh+15,
// with kh = (L>>4)*16 (contiguous, matches B-fragment order).
__device__ __forceinline__ v16bf load_a_frag(const __bf16* __restrict__ base,
                                             int stride, int mbase, int kb, int lane) {
  int row = mbase + (lane & 15);
  int kh  = (lane >> 4) * 16;
  const __bf16* p = base + (size_t)row * stride + kb + kh;
  BF16x16 u;
  u.h[0] = *(const v8bf*)p;
  u.h[1] = *(const v8bf*)(p + 8);
  return u.v;
}

// B-fragment (32x16, 16-bit): lane L holds column (L&15); K = (L>>4)*16 .. +15,
// packed sequentially. Source stored K-contiguous per column (i.e. transposed).
__device__ __forceinline__ v16bf load_b_frag(const __bf16* __restrict__ baseT,
                                             int strideK, int nbase, int kb, int lane) {
  const __bf16* p = baseT + (size_t)(nbase + (lane & 15)) * strideK + kb + (lane >> 4) * 16;
  return *(const v16bf*)p;
}

__device__ __forceinline__ float hmax16(float x) {
  x = fmaxf(x, __shfl_xor(x, 1, 16));
  x = fmaxf(x, __shfl_xor(x, 2, 16));
  x = fmaxf(x, __shfl_xor(x, 4, 16));
  x = fmaxf(x, __shfl_xor(x, 8, 16));
  return x;
}
__device__ __forceinline__ float hsum16(float x) {
  x += __shfl_xor(x, 1, 16);
  x += __shfl_xor(x, 2, 16);
  x += __shfl_xor(x, 4, 16);
  x += __shfl_xor(x, 8, 16);
  return x;
}

// ---------------- Elementwise helpers ----------------
__global__ void cvt_f32_to_bf16(const float* __restrict__ s, __bf16* __restrict__ d, int n) {
  int i = blockIdx.x * blockDim.x + threadIdx.x;
  if (i < n) d[i] = (__bf16)s[i];
}

// W (K x N, row major f32)  ->  Wt (N x K, row major bf16)
__global__ void transpose_to_bf16(const float* __restrict__ W, __bf16* __restrict__ Wt,
                                  int K, int N) {
  int i = blockIdx.x * blockDim.x + threadIdx.x;
  if (i >= K * N) return;
  int n = i % N, k = i / N;
  Wt[(size_t)n * K + k] = (__bf16)W[(size_t)k * N + n];
}

// ---------------- WMMA GEMM with async-LDS double-buffered B staging ----------------
// C(MxN,f32) = A(MxK,bf16) @ B(KxN given as Bt[N][K],bf16) + bias
// block = 256 threads = 8 waves; wave -> 16 rows x 64 cols; grid = (N/64, M/128).
// Per K-chunk (32), the 64x32 B tile (4 KB) is staged ONCE per workgroup into LDS via
// GLOBAL_LOAD_ASYNC_TO_LDS_B128 (double buffered), instead of 8x duplicated global loads.
#define BSTRIDE 40   // per-column LDS stride in elements (80 B): conflict-free, 16B-aligned
__global__ void gemm_wmma_bf16(const __bf16* __restrict__ A, const __bf16* __restrict__ Bt,
                               const float* __restrict__ bias, float* __restrict__ C,
                               int M, int N, int K) {
  __shared__ __align__(16) __bf16 lbuf[2][64 * BSTRIDE];   // 2 x 5 KB

  int tid   = threadIdx.x;
  int lane  = tid & 31;
  int wave  = tid >> 5;
  int mbase = (blockIdx.y * 8 + wave) * 16;
  int nbase = blockIdx.x * 64;

  // copy mapping: thread -> (column, 16B segment); 256 threads cover 64 cols x 64 B
  int cpcol = tid >> 2;            // 0..63
  int cpseg = (tid & 3) * 8;       // element offset 0,8,16,24 within the 32-elem chunk
  const __bf16* gsrc = Bt + (size_t)(nbase + cpcol) * K + cpseg;
  __bf16* ldst[2] = { &lbuf[0][cpcol * BSTRIDE + cpseg],
                      &lbuf[1][cpcol * BSTRIDE + cpseg] };

  stage16(gsrc, ldst[0]);          // prefetch chunk 0 into stage 0

  v8f acc[4] = {};
  int nch = K / 32;
  int ncol = lane & 15, khalf = (lane >> 4) * 16;
  for (int c = 0; c < nch; ++c) {
    wait_async_lds();
    __syncthreads();               // chunk c resident in lbuf[c&1]; prior reads retired
    if (c + 1 < nch) stage16(gsrc + (c + 1) * 32, ldst[(c + 1) & 1]);
    if (c + 2 < nch)               // hint the A stream ahead (global_prefetch_b8)
      __builtin_prefetch(A + (size_t)(mbase + ncol) * K + (c + 2) * 32, 0, 1);

    v16bf a = load_a_frag(A, K, mbase, c * 32, lane);
    const __bf16* lb = &lbuf[c & 1][0];
#pragma unroll
    for (int df = 0; df < 4; ++df) {
      v16bf b = *(const v16bf*)&lb[(df * 16 + ncol) * BSTRIDE + khalf];
      acc[df] = wmma_bf16(a, b, acc[df]);
    }
  }

  int rowoff = (lane >> 4) * 8;    // C/D layout: VGPR r -> row r (lanes 0-15) / r+8 (16-31)
#pragma unroll
  for (int df = 0; df < 4; ++df) {
    int col  = nbase + df * 16 + ncol;
    float bv = bias ? bias[col] : 0.0f;
#pragma unroll
    for (int r = 0; r < 8; ++r) {
      C[(size_t)(mbase + r + rowoff) * N + col] = acc[df][r] + bv;
    }
  }
}

// ---------------- RoPE + layout packing ----------------
__global__ void rope_pack_q(const float* __restrict__ Qf, const float* __restrict__ cs,
                            const float* __restrict__ sn, __bf16* __restrict__ Qo) {
  int i = blockIdx.x * blockDim.x + threadIdx.x;
  if (i >= BT_ * H_ * D_) return;
  int d   = i & 63;
  int h   = (i >> 6) % H_;
  int row = i / (H_ * D_);
  int t = row % T_, b = row / T_;
  const float* qr = Qf + (size_t)row * (H_ * D_) + h * D_;
  float v   = qr[d];
  float rot = (d < 32) ? -qr[d + 32] : qr[d - 32];
  float out = (v * cs[t * D_ + d] + rot * sn[t * D_ + d]) * 0.125f;   // 1/sqrt(64) folded
  Qo[((size_t)(b * H_ + h) * T_ + t) * D_ + d] = (__bf16)out;
}

__global__ void rope_pack_k(const float* __restrict__ Kf, const float* __restrict__ cs,
                            const float* __restrict__ sn, __bf16* __restrict__ Ko) {
  int i = blockIdx.x * blockDim.x + threadIdx.x;
  if (i >= BT_ * KV_ * D_) return;
  int d   = i & 63;
  int kv  = (i >> 6) % KV_;
  int row = i / (KV_ * D_);
  int t = row % T_, b = row / T_;
  const float* kr = Kf + (size_t)row * (KV_ * D_) + kv * D_;
  float v   = kr[d];
  float rot = (d < 32) ? -kr[d + 32] : kr[d - 32];
  float out = v * cs[t * D_ + d] + rot * sn[t * D_ + d];
  Ko[((size_t)(b * KV_ + kv) * T_ + t) * D_ + d] = (__bf16)out;
}

// V -> [b][kv][d][t] bf16 (transposed so PV B-fragments load contiguously over keys)
__global__ void pack_v(const float* __restrict__ Vf, __bf16* __restrict__ Vo) {
  int i = blockIdx.x * blockDim.x + threadIdx.x;
  if (i >= BT_ * KV_ * D_) return;
  int d   = i & 63;
  int kv  = (i >> 6) % KV_;
  int row = i / (KV_ * D_);
  int t = row % T_, b = row / T_;
  Vo[((size_t)(b * KV_ + kv) * D_ + d) * T_ + t] =
      (__bf16)Vf[(size_t)row * (KV_ * D_) + kv * D_ + d];
}

// ---------------- Flash attention (wave-per-16-query-tile, WMMA) ----------------
// grid = (B*H, T/128), block = 256 (8 waves). Online softmax, f32 state.
__global__ void flash_attn_wmma(const __bf16* __restrict__ Qb, const __bf16* __restrict__ Kb,
                                const __bf16* __restrict__ Vt, __bf16* __restrict__ Ob) {
  __shared__ __align__(16) __bf16 pbuf[8][16][40];   // per-wave P staging (C-layout -> A-layout)

  int lane = threadIdx.x & 31;
  int wave = threadIdx.x >> 5;
  int bh = blockIdx.x;
  int b = bh / H_, h = bh % H_;
  int kv = h / (H_ / KV_);                 // GQA: 7 Q heads per KV head
  int qb = blockIdx.y * 128 + wave * 16;   // first query row of this wave's tile

  const __bf16* Qh = Qb + (size_t)(b * H_ + h) * T_ * D_;
  const __bf16* Kh = Kb + (size_t)(b * KV_ + kv) * T_ * D_;
  const __bf16* Vh = Vt + (size_t)(b * KV_ + kv) * D_ * T_;

  // Q tile (16x64) resident as two A-fragments (scale 1/8 already folded in)
  v16bf qf[2];
#pragma unroll
  for (int c = 0; c < 2; ++c) qf[c] = load_a_frag(Qh, D_, qb, c * 32, lane);

  v8f o[4] = {};
  float m[8], l[8];
#pragma unroll
  for (int r = 0; r < 8; ++r) { m[r] = -1e30f; l[r] = 0.0f; }

  int rowoff = (lane >> 4) * 8;
  int ncol   = lane & 15;

  int ktend = (qb + 15) >> 5;              // causal: last 32-key tile touching this query tile
  for (int kt = 0; kt <= ktend; ++kt) {
    int k0 = kt * 32;
    if (kt < ktend) {                      // prefetch next K/V tiles (global_prefetch_b8)
      __builtin_prefetch(Kh + (size_t)(k0 + 32 + lane) * D_, 0, 1);
      __builtin_prefetch(Vh + (size_t)(lane * 2) * T_ + k0 + 32, 0, 1);
    }

    // ---- S = Q @ K^T  (16 x 32), two 16-col fragments, K-dim 64 in two chunks ----
    v8f s0 = {}, s1 = {};
#pragma unroll
    for (int c = 0; c < 2; ++c) {
      v16bf b0 = load_b_frag(Kh, D_, k0,      c * 32, lane);
      v16bf b1 = load_b_frag(Kh, D_, k0 + 16, c * 32, lane);
      s0 = wmma_bf16(qf[c], b0, s0);
      s1 = wmma_bf16(qf[c], b1, s1);
    }

    // ---- online softmax, per accumulator row slot ----
    bool needmask = (k0 + 31 > qb);
#pragma unroll
    for (int r = 0; r < 8; ++r) {
      float e0 = s0[r], e1 = s1[r];
      if (needmask) {
        int rowg = qb + r + rowoff;
        if (k0 + ncol > rowg)      e0 = -1e9f;   // NEG_INF, matches reference mask
        if (k0 + 16 + ncol > rowg) e1 = -1e9f;
      }
      float mx    = hmax16(fmaxf(e0, e1));
      float mnew  = fmaxf(m[r], mx);
      float alpha = __expf(m[r] - mnew);
      m[r] = mnew;
      e0 = __expf(e0 - mnew);
      e1 = __expf(e1 - mnew);
      float rs = hsum16(e0 + e1);
      l[r] = l[r] * alpha + rs;
#pragma unroll
      for (int df = 0; df < 4; ++df) o[df][r] = o[df][r] * alpha;
      // stage P (bf16) into LDS in row-major order for the A-fragment reload
      pbuf[wave][r + rowoff][ncol]      = (__bf16)e0;
      pbuf[wave][r + rowoff][16 + ncol] = (__bf16)e1;
    }
    asm volatile("s_wait_dscnt 0" ::: "memory");   // cross-lane LDS handoff within the wave

    // reload P as A-fragment (16 x 32)
    BF16x16 pa;
    {
      int prow = lane & 15;
      int kh   = (lane >> 4) * 16;
      pa.h[0] = *(const v8bf*)&pbuf[wave][prow][kh];
      pa.h[1] = *(const v8bf*)&pbuf[wave][prow][8 + kh];
    }

    // ---- O += P @ V  (16 x 64), four 16-col fragments, K-dim = 32 keys ----
#pragma unroll
    for (int df = 0; df < 4; ++df) {
      const __bf16* vp = Vh + (size_t)(df * 16 + ncol) * T_ + k0 + (lane >> 4) * 16;
      v16bf bv = *(const v16bf*)vp;
      o[df] = wmma_bf16(pa.v, bv, o[df]);
    }
    asm volatile("" ::: "memory");  // keep LDS reads ordered before next tile's writes
  }

  // ---- normalize and emit attention output (BT x 896, bf16) for the O-projection GEMM ----
#pragma unroll
  for (int r = 0; r < 8; ++r) {
    float inv = 1.0f / l[r];
    int row = qb + r + rowoff;
#pragma unroll
    for (int df = 0; df < 4; ++df) {
      Ob[(size_t)(b * T_ + row) * (H_ * D_) + h * D_ + df * 16 + ncol] =
          (__bf16)(o[df][r] * inv);
    }
  }
}

// ---------------- Host-side orchestration ----------------
extern "C" void kernel_launch(void* const* d_in, const int* in_sizes, int n_in,
                              void* d_out, int out_size, void* d_ws, size_t ws_size,
                              hipStream_t stream) {
  (void)in_sizes; (void)n_in; (void)out_size; (void)ws_size;

  const float* x       = (const float*)d_in[0];
  const float* cosb    = (const float*)d_in[1];
  const float* sinb    = (const float*)d_in[2];
  /* d_in[3] = mask: causal mask applied analytically in-kernel */
  const float* wq      = (const float*)d_in[4];
  const float* wq_bias = (const float*)d_in[5];
  const float* wk      = (const float*)d_in[6];
  const float* wk_bias = (const float*)d_in[7];
  const float* wv      = (const float*)d_in[8];
  const float* wv_bias = (const float*)d_in[9];
  const float* wo      = (const float*)d_in[10];
  float* out = (float*)d_out;

  // Workspace carve-up (deterministic each call)
  char* ws = (char*)d_ws;
  size_t off = 0;
  auto alloc = [&](size_t bytes) { char* p = ws + off; off += (bytes + 255) & ~(size_t)255; return p; };

  __bf16* xb   = (__bf16*)alloc((size_t)BT_ * DIM_ * 2);          // x in bf16
  __bf16* wqt  = (__bf16*)alloc((size_t)896 * DIM_ * 2);          // wq^T
  __bf16* wkt  = (__bf16*)alloc((size_t)128 * DIM_ * 2);          // wk^T
  __bf16* wvt  = (__bf16*)alloc((size_t)128 * DIM_ * 2);          // wv^T
  __bf16* wot  = (__bf16*)alloc((size_t)DIM_ * 896 * 2);          // wo^T
  float*  Qf   = (float*) alloc((size_t)BT_ * 896 * 4);           // Q pre-RoPE f32
  float*  Kf   = (float*) alloc((size_t)BT_ * 128 * 4);
  float*  Vf   = (float*) alloc((size_t)BT_ * 128 * 4);
  __bf16* Qb   = (__bf16*)alloc((size_t)BT_ * 896 * 2);           // [b][h][t][d]
  __bf16* Kb   = (__bf16*)alloc((size_t)BT_ * 128 * 2);           // [b][kv][t][d]
  __bf16* Vtb  = (__bf16*)alloc((size_t)BT_ * 128 * 2);           // [b][kv][d][t]
  __bf16* Ob   = (__bf16*)alloc((size_t)BT_ * 896 * 2);           // attn out bf16

  const int TPB = 256;

  // 1) precision conversion + weight transposes
  cvt_f32_to_bf16<<<(BT_ * DIM_ + TPB - 1) / TPB, TPB, 0, stream>>>(x, xb, BT_ * DIM_);
  transpose_to_bf16<<<(DIM_ * 896 + TPB - 1) / TPB, TPB, 0, stream>>>(wq, wqt, DIM_, 896);
  transpose_to_bf16<<<(DIM_ * 128 + TPB - 1) / TPB, TPB, 0, stream>>>(wk, wkt, DIM_, 128);
  transpose_to_bf16<<<(DIM_ * 128 + TPB - 1) / TPB, TPB, 0, stream>>>(wv, wvt, DIM_, 128);
  transpose_to_bf16<<<(896 * DIM_ + TPB - 1) / TPB, TPB, 0, stream>>>(wo, wot, 896, DIM_);

  // 2) QKV projections (WMMA bf16 -> f32, async-LDS staged weights)
  gemm_wmma_bf16<<<dim3(896 / 64, BT_ / 128), TPB, 0, stream>>>(xb, wqt, wq_bias, Qf, BT_, 896, DIM_);
  gemm_wmma_bf16<<<dim3(128 / 64, BT_ / 128), TPB, 0, stream>>>(xb, wkt, wk_bias, Kf, BT_, 128, DIM_);
  gemm_wmma_bf16<<<dim3(128 / 64, BT_ / 128), TPB, 0, stream>>>(xb, wvt, wv_bias, Vf, BT_, 128, DIM_);

  // 3) RoPE + attention-friendly layouts
  rope_pack_q<<<(BT_ * H_ * D_ + TPB - 1) / TPB, TPB, 0, stream>>>(Qf, cosb, sinb, Qb);
  rope_pack_k<<<(BT_ * KV_ * D_ + TPB - 1) / TPB, TPB, 0, stream>>>(Kf, cosb, sinb, Kb);
  pack_v<<<(BT_ * KV_ * D_ + TPB - 1) / TPB, TPB, 0, stream>>>(Vf, Vtb);

  // 4) fused causal flash attention (WMMA)
  flash_attn_wmma<<<dim3(B_ * H_, T_ / 128), TPB, 0, stream>>>(Qb, Kb, Vtb, Ob);

  // 5) output projection (WMMA bf16 -> f32 d_out)
  gemm_wmma_bf16<<<dim3(DIM_ / 64, BT_ / 128), TPB, 0, stream>>>(Ob, wot, nullptr, out, BT_, DIM_, 896);
}
